// LVCBlock_9345848836765
// MI455X (gfx1250) — matrix operand order repaired and checked by reading, not compiled, not run on hardware
//
#include <hip/hip_runtime.h>
#include <math.h>

typedef __attribute__((ext_vector_type(16))) __bf16          v16bf;
typedef __attribute__((ext_vector_type(8)))  float           v8f;
typedef __attribute__((ext_vector_type(8)))  unsigned short  v8us;

#define B_     4
#define CI_    32
#define CO_    64
#define CC_    100
#define L_     1024
#define T_     8192      // L * HOP
#define KH_    64
#define KD_    192       // GEMM K = 64 ch * 3 taps
#define MSLAB_ 6144      // per-layer out channels = 32*64*3
#define MTOT_  24576     // 4 layers * 6144

__device__ __forceinline__ float lrelu(float v) { return v >= 0.f ? v : 0.2f * v; }

__device__ __forceinline__ unsigned short f2bf(float f) {
    unsigned int u = __float_as_uint(f);
    u += 0x7FFFu + ((u >> 16) & 1u);          // round-to-nearest-even
    return (unsigned short)(u >> 16);
}
__device__ __forceinline__ float bf2f(unsigned short s) {
    return __uint_as_float(((unsigned int)s) << 16);
}

// ---------------------------------------------------------------------------
// 1. ConvTranspose1d (stride 8, K=16, pad 4), lrelu on input, bias on output.
//    y[b,o,t] = b_o + sum_ci sum_{k == (t+4) mod 8, k, k+8} w[ci,o,k]*lrelu(x[ci,(t+4-k)/8])
// ---------------------------------------------------------------------------
__global__ __launch_bounds__(256)
void convt_kernel(const float* __restrict__ x, const float* __restrict__ w,
                  const float* __restrict__ bias, float* __restrict__ y) {
    int gid = blockIdx.x * 256 + threadIdx.x;
    if (gid >= B_ * CI_ * T_) return;
    int t = gid & (T_ - 1);
    int o = (gid >> 13) & 31;
    int b = gid >> 18;
    float acc = bias[o];
    int r = (t + 4) & 7;
    #pragma unroll
    for (int kk = 0; kk < 2; ++kk) {
        int k = r + 8 * kk;
        int j = (t + 4 - k) >> 3;     // exact multiple of 8
        if (j >= 0 && j < L_) {
            #pragma unroll 4
            for (int ci = 0; ci < CI_; ++ci) {
                float v = x[(b * CI_ + ci) * L_ + j];
                acc += w[(ci * CI_ + o) * 16 + k] * lrelu(v);
            }
        }
    }
    y[gid] = acc;
}

// ---------------------------------------------------------------------------
// 2. Kernel-predictor input conv: c(B,100,1024) -> h(B,64,1024), K=5 pad2, lrelu
// ---------------------------------------------------------------------------
__global__ __launch_bounds__(256)
void kpin_kernel(const float* __restrict__ c, const float* __restrict__ w,
                 const float* __restrict__ bias, float* __restrict__ h) {
    int gid = blockIdx.x * 256 + threadIdx.x;
    if (gid >= B_ * KH_ * L_) return;
    int l  = gid & (L_ - 1);
    int oc = (gid >> 10) & 63;
    int b  = gid >> 16;
    float acc = bias[oc];
    for (int ci = 0; ci < CC_; ++ci) {
        const float* cp = c + (b * CC_ + ci) * L_;
        const float* wp = w + (oc * CC_ + ci) * 5;
        #pragma unroll
        for (int k = 0; k < 5; ++k) {
            int t = l + k - 2;
            if (t >= 0 && t < L_) acc += wp[k] * cp[t];
        }
    }
    h[gid] = lrelu(acc);
}

// ---------------------------------------------------------------------------
// 3. Generic 64-ch-in conv, K=3 pad1; optional lrelu; optional residual add.
//    Used for the 3 residual blocks (OC=64) and the bias head (OC=256).
// ---------------------------------------------------------------------------
__global__ __launch_bounds__(256)
void conv64_kernel(const float* __restrict__ in, const float* __restrict__ w,
                   const float* __restrict__ bias, const float* addsrc,
                   float* out, int OC, int do_lrelu) {
    int gid = blockIdx.x * 256 + threadIdx.x;
    if (gid >= B_ * OC * L_) return;
    int l  = gid & (L_ - 1);
    int oc = (gid >> 10) % OC;
    int b  = gid / (OC * L_);
    float acc = bias[oc];
    const float* ip = in + (size_t)b * KH_ * L_;
    const float* wp = w + (size_t)oc * KH_ * 3;
    for (int ci = 0; ci < KH_; ++ci) {
        #pragma unroll
        for (int k = 0; k < 3; ++k) {
            int t = l + k - 1;
            if (t >= 0 && t < L_) acc += wp[ci * 3 + k] * ip[ci * L_ + t];
        }
    }
    if (do_lrelu) acc = lrelu(acc);
    if (addsrc) acc += addsrc[gid];
    out[gid] = acc;
}

// ---------------------------------------------------------------------------
// 4. fp32 -> bf16 bulk convert (weight matrix)
// ---------------------------------------------------------------------------
__global__ __launch_bounds__(256)
void f2bf_kernel(const float* __restrict__ src, unsigned short* __restrict__ dst, int n) {
    int gid = blockIdx.x * 256 + threadIdx.x;
    if (gid < n) dst[gid] = f2bf(src[gid]);
}

// ---------------------------------------------------------------------------
// 5. im2col pack of h into bf16 B-matrix: Hb[(b*L+l)*192 + c*3+k] = h[b,c,l+k-1]
// ---------------------------------------------------------------------------
__global__ __launch_bounds__(256)
void hpack_kernel(const float* __restrict__ h, unsigned short* __restrict__ Hb) {
    int gid = blockIdx.x * 256 + threadIdx.x;
    if (gid >= B_ * L_ * KD_) return;
    int kd = gid % KD_;
    int n  = gid / KD_;
    int c  = kd / 3, k = kd % 3;
    int l  = n & (L_ - 1);
    int b  = n >> 10;
    int t  = l + k - 1;
    float v = (t >= 0 && t < L_) ? h[(b * KH_ + c) * L_ + t] : 0.f;
    Hb[gid] = f2bf(v);
}

// ---------------------------------------------------------------------------
// 6. Dilated 32x32 conv block: out = lrelu(conv(lrelu(x), K=3, dil, pad=dil))
// ---------------------------------------------------------------------------
__global__ __launch_bounds__(256)
void cbconv_kernel(const float* __restrict__ x, const float* __restrict__ w,
                   const float* __restrict__ bias, float* __restrict__ out, int dil) {
    int gid = blockIdx.x * 256 + threadIdx.x;
    if (gid >= B_ * CI_ * T_) return;
    int t  = gid & (T_ - 1);
    int co = (gid >> 13) & 31;
    int b  = gid >> 18;
    float acc = bias[co];
    const float* xp = x + (size_t)b * CI_ * T_;
    for (int ci = 0; ci < CI_; ++ci) {
        const float* wp = w + (co * CI_ + ci) * 3;
        #pragma unroll
        for (int k = 0; k < 3; ++k) {
            int tt = t + (k - 1) * dil;
            if (tt >= 0 && tt < T_) acc += wp[k] * lrelu(xp[ci * T_ + tt]);
        }
    }
    out[gid] = lrelu(acc);
}

// ---------------------------------------------------------------------------
// 7. WMMA bf16 GEMM: kf[n, m] = bias[m] + sum_kd Wb[m, kd] * Hb[n, kd]
//    One wave -> one 16x16 D tile, 6x v_wmma_f32_16x16x32_bf16 over K=192.
//    A lane (row m): K chunks [h*8,h*8+8) and [16+h*8,...)  -> two b128 loads
//    B lane (col n): K chunks [h*8,h*8+8) and [16+h*8,...)               -> two b128 loads
//    D lane (col n): rows m0+8h..m0+8h+7 contiguous         -> one  b128 store
//    Per-layer-slab output: kf is [1024, 6144] bf16 for one (layer, batch).
// ---------------------------------------------------------------------------
__global__ __launch_bounds__(256)
void lvc_gemm_wmma(const unsigned short* __restrict__ Wb,   // [6144,192] (layer slab)
                   const unsigned short* __restrict__ Hb,   // [1024,192] (batch slab)
                   const float* __restrict__ bias,          // [6144]
                   unsigned short* __restrict__ kf) {       // [1024,6144]
    const int lane = threadIdx.x & 31;
    const int wave = threadIdx.x >> 5;                       // 8 waves
    const int m0   = blockIdx.x * 64 + (wave & 3) * 16;
    const int n0   = blockIdx.y * 32 + (wave >> 2) * 16;
    const int mn   = lane & 15;
    const int half = lane >> 4;

    v8f acc = {0.f, 0.f, 0.f, 0.f, 0.f, 0.f, 0.f, 0.f};
    const unsigned short* arow = Wb + (size_t)(m0 + mn) * KD_;
    const unsigned short* bcol = Hb + (size_t)(n0 + mn) * KD_;

    union AB { v16bf v; v8us h[2]; };
    #pragma unroll
    for (int kk = 0; kk < KD_; kk += 32) {
        AB A, Bm;
        A.h[0]  = *(const v8us*)(arow + kk + half * 8);       // K = h*8 .. +8
        A.h[1]  = *(const v8us*)(arow + kk + 16 + half * 8);  // K = 16+h*8 ..
        Bm.h[0] = *(const v8us*)(bcol + kk + half * 8);       // K = h*8 .. +8
        Bm.h[1] = *(const v8us*)(bcol + kk + 16 + half * 8);
        acc = __builtin_amdgcn_wmma_f32_16x16x32_bf16(
            false, A.v, false, Bm.v, (short)0, acc, false, false);
    }

    int mb = m0 + half * 8;
    v8us ov;
    #pragma unroll
    for (int r = 0; r < 8; ++r) ov[r] = f2bf(acc[r] + bias[mb + r]);
    *(v8us*)(kf + (size_t)(n0 + mn) * MSLAB_ + mb) = ov;
}

// ---------------------------------------------------------------------------
// 8. LVC apply (one batch): o[b,co,l,s] = biases[b,layer,co,l]
//        + sum_{ci,k} conv_out[b,ci, l*8+s+k-1] * kf[l, ci*192+co*3+k]
// ---------------------------------------------------------------------------
__global__ __launch_bounds__(256)
void lvc_apply_kernel(const unsigned short* __restrict__ kf,   // [1024,6144]
                      const float* __restrict__ convout,       // [B,32,8192]
                      const float* __restrict__ biasbuf,       // [B,256,1024]
                      int layer, int b, float* __restrict__ obuf) {
    int gid = blockIdx.x * 256 + threadIdx.x;                  // CO*L*8 = 524288
    if (gid >= CO_ * L_ * 8) return;
    int s  = gid & 7;
    int l  = (gid >> 3) & (L_ - 1);
    int co = gid >> 13;
    float acc = biasbuf[((b * 256) + layer * 64 + co) * L_ + l];
    const unsigned short* kp = kf + (size_t)l * MSLAB_ + co * 3;
    const float* cp = convout + (size_t)b * CI_ * T_;
    int tb = l * 8 + s - 1;
    for (int ci = 0; ci < CI_; ++ci) {
        const unsigned short* kc = kp + ci * KD_;
        const float* cc = cp + ci * T_;
        #pragma unroll
        for (int k = 0; k < 3; ++k) {
            int tt = tb + k;
            float xv = (tt >= 0 && tt < T_) ? cc[tt] : 0.f;
            acc += bf2f(kc[k]) * xv;
        }
    }
    obuf[((size_t)b * CO_ + co) * T_ + l * 8 + s] = acc;
}

// ---------------------------------------------------------------------------
// 9. Gated update: x += sigmoid(o[:, :32]) * tanh(o[:, 32:])
// ---------------------------------------------------------------------------
__global__ __launch_bounds__(256)
void gate_update_kernel(const float* __restrict__ obuf, float* __restrict__ x) {
    int gid = blockIdx.x * 256 + threadIdx.x;
    if (gid >= B_ * CI_ * T_) return;
    int t = gid & (T_ - 1);
    int c = (gid >> 13) & 31;
    int b = gid >> 18;
    float a = obuf[((size_t)b * CO_ + c) * T_ + t];
    float g = obuf[((size_t)b * CO_ + 32 + c) * T_ + t];
    float sig = 1.f / (1.f + __expf(-a));
    x[gid] += sig * tanhf(g);
}

// ---------------------------------------------------------------------------
extern "C" void kernel_launch(void* const* d_in, const int* in_sizes, int n_in,
                              void* d_out, int out_size, void* d_ws, size_t ws_size,
                              hipStream_t stream) {
    const float* x_in    = (const float*)d_in[0];
    const float* c_in    = (const float*)d_in[1];
    const float* convt_w = (const float*)d_in[2];
    const float* convt_b = (const float*)d_in[3];
    const float* kpin_w  = (const float*)d_in[4];
    const float* kpin_b  = (const float*)d_in[5];
    const float* res_w   = (const float*)d_in[6];
    const float* res_b   = (const float*)d_in[7];
    const float* kk_w    = (const float*)d_in[8];
    const float* kk_b    = (const float*)d_in[9];
    const float* kb_w    = (const float*)d_in[10];
    const float* kb_b    = (const float*)d_in[11];
    const float* cb_w    = (const float*)d_in[12];
    const float* cb_b    = (const float*)d_in[13];
    float* x = (float*)d_out;                     // (B, 32, 8192) lives in d_out

    // workspace carve (256-B aligned slabs), total ~41 MiB
    char* p = (char*)d_ws;
    auto carve = [&](size_t bytes) -> char* {
        char* r = p; p += (bytes + 255) & ~(size_t)255; return r;
    };
    float*          h_buf    = (float*)carve(sizeof(float) * B_ * KH_ * L_);           // 1 MiB
    float*          t_buf    = (float*)carve(sizeof(float) * B_ * KH_ * L_);           // 1 MiB
    float*          bias_buf = (float*)carve(sizeof(float) * B_ * 256 * L_);           // 4 MiB
    float*          conv_out = (float*)carve(sizeof(float) * B_ * CI_ * T_);           // 4 MiB
    float*          obuf     = (float*)carve(sizeof(float) * B_ * CO_ * T_);           // 8 MiB
    unsigned short* Wb       = (unsigned short*)carve(sizeof(unsigned short) * (size_t)MTOT_ * KD_); // 9 MiB
    unsigned short* Hb       = (unsigned short*)carve(sizeof(unsigned short) * (size_t)B_ * L_ * KD_); // 1.5 MiB
    unsigned short* kf       = (unsigned short*)carve(sizeof(unsigned short) * (size_t)L_ * MSLAB_);   // 12 MiB

    const int TB = 256;
    auto blks = [](int n) { return (n + 255) / 256; };

    // 1. upsample x -> d_out
    convt_kernel<<<blks(B_ * CI_ * T_), TB, 0, stream>>>(x_in, convt_w, convt_b, x);
    // 2. kernel predictor frontend
    kpin_kernel<<<blks(B_ * KH_ * L_), TB, 0, stream>>>(c_in, kpin_w, kpin_b, h_buf);
    // 3. three residual blocks (h = h + lrelu(conv(lrelu(conv(h)))))
    for (int i = 0; i < 3; ++i) {
        conv64_kernel<<<blks(B_ * KH_ * L_), TB, 0, stream>>>(
            h_buf, res_w + (size_t)(i * 2 + 0) * KH_ * KH_ * 3, res_b + (i * 2 + 0) * KH_,
            nullptr, t_buf, KH_, 1);
        conv64_kernel<<<blks(B_ * KH_ * L_), TB, 0, stream>>>(
            t_buf, res_w + (size_t)(i * 2 + 1) * KH_ * KH_ * 3, res_b + (i * 2 + 1) * KH_,
            h_buf, h_buf, KH_, 1);
    }
    // 4. bias head (all 4 layers, 256 channels)
    conv64_kernel<<<blks(B_ * 256 * L_), TB, 0, stream>>>(
        h_buf, kb_w, kb_b, nullptr, bias_buf, 256, 0);
    // 5. weight matrix -> bf16 (layout already [m, c*3+k])
    f2bf_kernel<<<blks(MTOT_ * KD_), TB, 0, stream>>>(kk_w, Wb, MTOT_ * KD_);
    // 6. im2col pack of h -> bf16 B matrix
    hpack_kernel<<<blks(B_ * L_ * KD_), TB, 0, stream>>>(h_buf, Hb);

    // 7. LVC layers
    const int dil[4] = {1, 3, 9, 27};
    for (int i = 0; i < 4; ++i) {
        cbconv_kernel<<<blks(B_ * CI_ * T_), TB, 0, stream>>>(
            x, cb_w + (size_t)i * CI_ * CI_ * 3, cb_b + i * CI_, conv_out, dil[i]);
        for (int b = 0; b < B_; ++b) {
            lvc_gemm_wmma<<<dim3(MSLAB_ / 64, L_ / 32), dim3(256), 0, stream>>>(
                Wb + (size_t)i * MSLAB_ * KD_, Hb + (size_t)b * L_ * KD_,
                kk_b + (size_t)i * MSLAB_, kf);
            lvc_apply_kernel<<<blks(CO_ * L_ * 8), TB, 0, stream>>>(
                kf, conv_out, bias_buf, i, b, obuf);
        }
        gate_update_kernel<<<blks(B_ * CI_ * T_), TB, 0, stream>>>(obuf, x);
    }
}
